// QFCModel_65481071406192
// MI455X (gfx1250) — compile-verified, matrix operand order checked by reading, not run on hardware
//
#include <hip/hip_runtime.h>
#include <hip/hip_bf16.h>

typedef __attribute__((ext_vector_type(2)))  float    v2f;
typedef __attribute__((ext_vector_type(8)))  float    v8f;
typedef __attribute__((ext_vector_type(16))) _Float16 v16h;

// ---------------------------------------------------------------------------
// Kernel 1: 6x6 avg-pool of x[:, :, :24, :24] -> pooled[B,16]
// one thread per pooled cell; 3 x float2 loads per row (8B aligned: all index
// components even), coalesced across the 16 cells of neighboring samples.
// ---------------------------------------------------------------------------
__global__ __launch_bounds__(256) void pool_kernel(const float* __restrict__ x,
                                                   float* __restrict__ pooled,
                                                   int total) {
  int t = blockIdx.x * blockDim.x + threadIdx.x;
  if (t >= total) return;
  int s = t >> 4, cell = t & 15;
  int py = cell >> 2, px = cell & 3;
  const float* p = x + (size_t)s * 784 + py * 6 * 28 + px * 6;
  float sum = 0.f;
#pragma unroll
  for (int r = 0; r < 6; ++r) {
    const float2* q = (const float2*)(p + r * 28);
    float2 v0 = q[0], v1 = q[1], v2 = q[2];
    sum += (v0.x + v0.y) + (v1.x + v1.y) + (v2.x + v2.y);
  }
  pooled[t] = sum * (1.f / 36.f);
}

// ---------------------------------------------------------------------------
// Statevector gates: 4 wires, 16 complex amplitudes in registers.
// Wire w acts on bit of weight (8 >> w)  [matches reference reshape order].
// Fully unrolled: stride is a compile-time constant at every call site.
// ---------------------------------------------------------------------------
__device__ __forceinline__ void gate_rx(float re[16], float im[16], int stride, float t) {
  float c = cosf(0.5f * t), s = sinf(0.5f * t);
#pragma unroll
  for (int i = 0; i < 16; ++i) {
    if (i & stride) continue;
    int j = i | stride;
    float r0 = re[i], i0 = im[i], r1 = re[j], i1 = im[j];
    re[i] = c * r0 + s * i1;   // c*a0 - i*s*a1
    im[i] = c * i0 - s * r1;
    re[j] = c * r1 + s * i0;   // -i*s*a0 + c*a1
    im[j] = c * i1 - s * r0;
  }
}
__device__ __forceinline__ void gate_ry(float re[16], float im[16], int stride, float t) {
  float c = cosf(0.5f * t), s = sinf(0.5f * t);
#pragma unroll
  for (int i = 0; i < 16; ++i) {
    if (i & stride) continue;
    int j = i | stride;
    float r0 = re[i], i0 = im[i], r1 = re[j], i1 = im[j];
    re[i] = c * r0 - s * r1;
    im[i] = c * i0 - s * i1;
    re[j] = s * r0 + c * r1;
    im[j] = s * i0 + c * i1;
  }
}
__device__ __forceinline__ void gate_rz(float re[16], float im[16], int stride, float t) {
  float c = cosf(0.5f * t), s = sinf(0.5f * t);
#pragma unroll
  for (int i = 0; i < 16; ++i) {
    if (i & stride) continue;
    int j = i | stride;
    float r0 = re[i], i0 = im[i], r1 = re[j], i1 = im[j];
    re[i] = c * r0 + s * i0;   // (c - i*s)*a0
    im[i] = c * i0 - s * r0;
    re[j] = c * r1 - s * i1;   // (c + i*s)*a1
    im[j] = c * i1 + s * r1;
  }
}
__device__ __forceinline__ void gate_cnot(float re[16], float im[16], int cw, int tw) {
#pragma unroll
  for (int i = 0; i < 16; ++i) {
    if ((i & cw) && !(i & tw)) {
      int j = i | tw;
      float tr = re[i]; re[i] = re[j]; re[j] = tr;
      float ti = im[i]; im[i] = im[j]; im[j] = ti;
    }
  }
}

// ---------------------------------------------------------------------------
// Kernel 2: feats = pooled @ W + b via V_WMMA_F32_16X16X4_F32 (two 16-sample
// tiles per wave, K=16 as 4 steps of K=4), then per-lane circuit simulation,
// raw out -> d_out, deterministic block partial sums -> ws.
// B-fragment built branchlessly: unconditional loads from clamped W address,
// zero-masked for columns >= 4 (keeps EXEC untouched, no saveexec blocks).
// ---------------------------------------------------------------------------
__global__ __launch_bounds__(256) void circuit_kernel(const float* __restrict__ pooled,
                                                      const float* __restrict__ W,
                                                      const float* __restrict__ bvec,
                                                      const float* __restrict__ vp,
                                                      float* __restrict__ outraw,
                                                      float* __restrict__ partials) {
  __shared__ float lfeats[8 * 32 * 4];  // per wave: 32 samples x 4 channels
  __shared__ float wsum[8][8];

  const int tid  = threadIdx.x;
  const int wave = tid >> 5;
  const int lane = tid & 31;
  const int s0   = blockIdx.x * 256 + wave * 32;   // first of this wave's 32 samples
  const int m    = lane & 15;                      // row (sample within tile)
  const int kh   = lane >> 4;                      // lane-half selects K pair
  const int n    = m;                              // C/D column index for this lane

  // Hoisted, branchless B fragments: W[k][n] for k = kk*4 + kh*2 + {0,1}.
  const int   nc    = n & 3;                       // clamped column (always in-bounds)
  const float wmask = (n < 4) ? 1.f : 0.f;         // one v_cndmask, reused 8x
  v2f bfrag[4];
#pragma unroll
  for (int kk = 0; kk < 4; ++kk) {
    const int k0 = kk * 4 + kh * 2;
    bfrag[kk].x = W[(k0 + 0) * 4 + nc] * wmask;
    bfrag[kk].y = W[(k0 + 1) * 4 + nc] * wmask;
  }

  v8f c0 = {}; v8f c1 = {};
#if __has_builtin(__builtin_amdgcn_wmma_f32_16x16x4_f32)
#pragma unroll
  for (int kk = 0; kk < 4; ++kk) {
    const int k0 = kk * 4 + kh * 2;                // A fragment K base for this lane-half
    v2f a0 = *(const v2f*)(pooled + (size_t)(s0 + m) * 16 + k0);
    v2f a1 = *(const v2f*)(pooled + (size_t)(s0 + 16 + m) * 16 + k0);
    c0 = __builtin_amdgcn_wmma_f32_16x16x4_f32(false, a0, false, bfrag[kk], (short)0, c0, false, false);
    c1 = __builtin_amdgcn_wmma_f32_16x16x4_f32(false, a1, false, bfrag[kk], (short)0, c1, false, false);
  }
#else
  // Fallback: codegen-confirmed f16 WMMA, K=32 padded (valid K = 0..15).
  v16h af0 = {}, af1 = {}, bf16 = {};
#pragma unroll
  for (int e = 0; e < 8; ++e) {                    // e>=8 maps to K>=16 -> stays zero
    int kA = e + 8 * kh;                           // A: lanes0-15 K=0..7, lanes16-31 K=8..15
    af0[e] = (_Float16)pooled[(size_t)(s0 + m) * 16 + kA];
    af1[e] = (_Float16)pooled[(size_t)(s0 + 16 + m) * 16 + kA];
  }
  const float bm16 = (kh == 0) ? wmask : 0.f;
#pragma unroll
  for (int e = 0; e < 16; ++e) {                   // B: lanes0-15 K=0..15; lanes16-31 K=16..31 (zero)
    bf16[e] = (_Float16)(W[e * 4 + nc] * bm16);
  }
  c0 = __builtin_amdgcn_wmma_f32_16x16x32_f16(false, af0, false, bf16, (short)0, c0, false, false);
  c1 = __builtin_amdgcn_wmma_f32_16x16x32_f16(false, af1, false, bf16, (short)0, c1, false, false);
#endif

  // C layout: VGPR r holds row (r + 8*kh), column = lane&15. Stage cols 0..3.
  float* lf = lfeats + wave * 128;
  if (n < 4) {
#pragma unroll
    for (int r = 0; r < 8; ++r) {
      lf[(r + 8 * kh) * 4 + n]      = c0[r];
      lf[(16 + r + 8 * kh) * 4 + n] = c1[r];
    }
  }
  __syncthreads();

  // Each lane simulates its own sample: s = s0 + lane.
  float th[4];
#pragma unroll
  for (int c = 0; c < 4; ++c) th[c] = lf[lane * 4 + c] + bvec[c];

  float re[16], im[16];
#pragma unroll
  for (int i = 0; i < 16; ++i) { re[i] = 0.f; im[i] = 0.f; }
  re[0] = 1.f;

#pragma unroll
  for (int w = 0; w < 4; ++w) gate_rx(re, im, 8 >> w, th[w]);   // encoding
#pragma unroll
  for (int l = 0; l < 3; ++l) {
#pragma unroll
    for (int w = 0; w < 4; ++w) {
      const float* g = vp + (l * 4 + w) * 3;                    // uniform -> s_load
      gate_rx(re, im, 8 >> w, g[0]);
      gate_ry(re, im, 8 >> w, g[1]);
      gate_rz(re, im, 8 >> w, g[2]);
    }
#pragma unroll
    for (int w = 0; w < 3; ++w) gate_cnot(re, im, 8 >> w, 4 >> w);
  }

  float o[4];
#pragma unroll
  for (int w = 0; w < 4; ++w) {
    const int mask = 8 >> w;
    float acc = 0.f;
#pragma unroll
    for (int i = 0; i < 16; ++i) {
      float p = re[i] * re[i] + im[i] * im[i];
      acc += (i & mask) ? -p : p;
    }
    o[w] = acc;
  }

  const int s = s0 + lane;
  *(float4*)(outraw + (size_t)s * 4) = make_float4(o[0], o[1], o[2], o[3]);

  // Deterministic reduction: wave32 xor-shuffle, then fixed-order across 8 waves.
  float a8[8];
#pragma unroll
  for (int c = 0; c < 4; ++c) { a8[c] = o[c]; a8[4 + c] = o[c] * o[c]; }
#pragma unroll
  for (int off = 16; off >= 1; off >>= 1) {
#pragma unroll
    for (int c = 0; c < 8; ++c) a8[c] += __shfl_xor(a8[c], off, 32);
  }
  if (lane == 0) {
#pragma unroll
    for (int c = 0; c < 8; ++c) wsum[wave][c] = a8[c];
  }
  __syncthreads();
  if (tid < 8) {
    float ssum = 0.f;
#pragma unroll
    for (int w = 0; w < 8; ++w) ssum += wsum[w][tid];
    partials[(size_t)blockIdx.x * 8 + tid] = ssum;
  }
}

// ---------------------------------------------------------------------------
// Kernel 3: reduce per-block partials (fixed tree order -> deterministic),
// produce per-channel scale/shift for the BatchNorm affine.
// ---------------------------------------------------------------------------
__global__ __launch_bounds__(256) void stats_kernel(const float* __restrict__ partials,
                                                    const float* __restrict__ gamma,
                                                    const float* __restrict__ beta,
                                                    float* __restrict__ stats,
                                                    int nblocks, float invB) {
  __shared__ float red[256];
  __shared__ float tot[8];
  const int t = threadIdx.x;
  for (int c = 0; c < 8; ++c) {
    red[t] = (t < nblocks) ? partials[(size_t)t * 8 + c] : 0.f;
    __syncthreads();
    for (int off = 128; off >= 1; off >>= 1) {
      if (t < off) red[t] += red[t + off];
      __syncthreads();
    }
    if (t == 0) tot[c] = red[0];
    __syncthreads();
  }
  if (t == 0) {
#pragma unroll
    for (int c = 0; c < 4; ++c) {
      float mu  = tot[c] * invB;
      float var = tot[4 + c] * invB - mu * mu;
      float sc  = gamma[c] * rsqrtf(var + 1e-5f);
      stats[c]     = sc;
      stats[4 + c] = beta[c] - mu * sc;
    }
  }
}

// ---------------------------------------------------------------------------
// Kernel 4: in-place elementwise BN affine, one float4 (one sample) per thread.
// ---------------------------------------------------------------------------
__global__ __launch_bounds__(256) void bn_kernel(float* __restrict__ out,
                                                 const float* __restrict__ stats,
                                                 int nsamples) {
  int s = blockIdx.x * blockDim.x + threadIdx.x;
  if (s >= nsamples) return;
  float4 v = *(float4*)(out + (size_t)s * 4);
  v.x = v.x * stats[0] + stats[4];
  v.y = v.y * stats[1] + stats[5];
  v.z = v.z * stats[2] + stats[6];
  v.w = v.w * stats[3] + stats[7];
  *(float4*)(out + (size_t)s * 4) = v;
}

// ---------------------------------------------------------------------------
extern "C" void kernel_launch(void* const* d_in, const int* in_sizes, int n_in,
                              void* d_out, int out_size, void* d_ws, size_t ws_size,
                              hipStream_t stream) {
  const float* x     = (const float*)d_in[0];  // [B,1,28,28]
  const float* W     = (const float*)d_in[1];  // [16,4]
  const float* bvec  = (const float*)d_in[2];  // [4]
  const float* vp    = (const float*)d_in[3];  // [3,4,3]
  const float* gamma = (const float*)d_in[4];  // [4]
  const float* beta  = (const float*)d_in[5];  // [4]
  float* out = (float*)d_out;                  // [B,4]

  const int B = in_sizes[0] / 784;             // 65536
  const int nblocks = B / 256;                 // 256 circuit blocks

  float* pooled   = (float*)d_ws;              // B*16 floats (4 MB)
  float* partials = pooled + (size_t)B * 16;   // nblocks*8 floats
  float* stats    = partials + (size_t)nblocks * 8;  // 8 floats (scale[4], shift[4])

  pool_kernel<<<(B * 16 + 255) / 256, 256, 0, stream>>>(x, pooled, B * 16);
  circuit_kernel<<<nblocks, 256, 0, stream>>>(pooled, W, bvec, vp, out, partials);
  stats_kernel<<<1, 256, 0, stream>>>(partials, gamma, beta, stats, nblocks, 1.f / (float)B);
  bn_kernel<<<(B + 255) / 256, 256, 0, stream>>>(out, stats, B);
}
